// PersistentHomologyLayer_25872882991125
// MI455X (gfx1250) — matrix operand order repaired and checked by reading, not compiled, not run on hardware
//
#include <hip/hip_runtime.h>

typedef __attribute__((ext_vector_type(16))) _Float16 v16h;
typedef __attribute__((ext_vector_type(8)))  float    v8f;

#define B_BATCH 65536
#define NPTS 14
#define NM1 13          // dim-0 features per batch
#define NF1 3           // dim-1 features per batch
#define FOUT 512

// ---- workspace layout (float offsets) ----
#define OFF_DEATHS0 0ul                       // B*13
#define OFF_B1      851968ul                  // B*3  births1
#define OFF_D1      1048576ul                 // B*3  deaths1
#define OFF_P0      1245184ul                 // B    deaths0_sorted[1]-[0]
#define OFF_Q1      1310720ul                 // B    births1[1]-births1[0]
#define OFF_Q2      1376256ul                 // B    deaths1[1]-deaths1[0]
#define OFF_ACC0    1441792ul                 // 32   sum of h2 over diag0 rows
#define OFF_ACC1    1441824ul                 // 32   sum of h2 over diag1 rows
#define OFF_S       1441856ul                 // 2    entropy normalizers S0,S1
#define OFF_E       1441858ul                 // 1    entropy accumulator
// total ~1.44M floats = 5.8 MB of d_ws

// ---------------------------------------------------------------------------
// Zero the accumulator region each call (harness poisons d_ws once; we must
// re-zero every call anyway for deterministic replay).
__global__ void zero_kernel(float* __restrict__ ws) {
  int t = threadIdx.x;
  if (t < 67) ws[OFF_ACC0 + t] = 0.0f;
}

// ---------------------------------------------------------------------------
// Geometry: one wave32 per batch. Distance matrix in LDS, Prim's MST with
// wave-wide argmin (shfl_xor -> ds_permute on gfx1250), dmax reduction,
// dim-1 births/deaths, entropy gap stats, euler constant.
__global__ __launch_bounds__(256) void geom_kernel(
    const float* __restrict__ pc, const float* __restrict__ rb,
    const float* __restrict__ rd, float* __restrict__ ws,
    float* __restrict__ out_euler) {
  __shared__ float s_pts[8][48];
  __shared__ float s_D[8][NPTS * NPTS];
  __shared__ float s_S[2];
  const int w = threadIdx.x >> 5;
  const int lane = threadIdx.x & 31;
  const int batch = blockIdx.x * 8 + w;
  if (threadIdx.x == 0) { s_S[0] = 0.0f; s_S[1] = 0.0f; }

  for (int t = lane; t < NPTS * 3; t += 32)
    s_pts[w][t] = pc[(size_t)batch * (NPTS * 3) + t];
  __syncthreads();

  // pairwise distances, lane-parallel over the 196 entries
  float lmax = 0.0f;
  for (int e = lane; e < NPTS * NPTS; e += 32) {
    int i = e / NPTS, j = e % NPTS;
    float dx = s_pts[w][i * 3 + 0] - s_pts[w][j * 3 + 0];
    float dy = s_pts[w][i * 3 + 1] - s_pts[w][j * 3 + 1];
    float dz = s_pts[w][i * 3 + 2] - s_pts[w][j * 3 + 2];
    float d2 = dx * dx + dy * dy + dz * dz;
    float d = sqrtf(fmaxf(d2, 1e-24f));
    s_D[w][e] = d;
    lmax = fmaxf(lmax, d);
  }
  __syncthreads();
  for (int off = 16; off; off >>= 1) lmax = fmaxf(lmax, __shfl_xor(lmax, off));
  const float dmax = lmax;

  // Prim's MST: lane j owns mind[j]; 13 serial steps with wave-wide argmin
  bool in_tree = (lane == 0);
  float mind = (lane >= 1 && lane < NPTS) ? s_D[w][lane] : 1e30f;
  float m1 = 1e30f, m2 = 1e30f;  // two smallest MST edge weights
  for (int step = 0; step < NM1; ++step) {
    float v = mind;
    int idx = lane;
    for (int off = 16; off; off >>= 1) {
      float v2 = __shfl_xor(v, off);
      int i2 = __shfl_xor(idx, off);
      if (v2 < v || (v2 == v && i2 < idx)) { v = v2; idx = i2; }
    }
    if (lane == 0) ws[OFF_DEATHS0 + (size_t)batch * NM1 + step] = v;
    if (v < m1) { m2 = m1; m1 = v; } else if (v < m2) { m2 = v; }
    if (lane == idx) in_tree = true;
    float dj = (lane < NPTS) ? s_D[w][idx * NPTS + lane] : 1e30f;
    mind = in_tree ? 1e30f : fminf(mind, dj);
  }

  // dim-1 diagram (randomized births/deaths scaled by dmax)
  if (lane < NF1) {
    float bb = rb[(size_t)batch * NF1 + lane] * dmax * 0.3f;
    float dd = bb + rd[(size_t)batch * NF1 + lane] * dmax * 0.4f;
    ws[OFF_B1 + (size_t)batch * NF1 + lane] = bb;
    ws[OFF_D1 + (size_t)batch * NF1 + lane] = dd;
  }
  if (lane == 0) {
    float bb0 = rb[(size_t)batch * NF1 + 0] * dmax * 0.3f;
    float bb1 = rb[(size_t)batch * NF1 + 1] * dmax * 0.3f;
    float dd0 = bb0 + rd[(size_t)batch * NF1 + 0] * dmax * 0.4f;
    float dd1 = bb1 + rd[(size_t)batch * NF1 + 1] * dmax * 0.4f;
    float p0 = m2 - m1, q1 = bb1 - bb0, q2 = dd1 - dd0;
    ws[OFF_P0 + batch] = p0;
    ws[OFF_Q1 + batch] = q1;
    ws[OFF_Q2 + batch] = q2;
    atomicAdd(&s_S[0], p0 > 0.0f ? p0 : 0.0f);
    atomicAdd(&s_S[1], (q1 > 0.0f ? q1 : 0.0f) + (q2 > 0.0f ? q2 : 0.0f));
    out_euler[batch] = (float)(NM1 - NF1);  // euler = b0 - b1 = 10
  }
  __syncthreads();
  if (threadIdx.x == 0) {
    atomicAdd(&ws[OFF_S + 0], s_S[0]);
    atomicAdd(&ws[OFF_S + 1], s_S[1]);
  }
}

// ---------------------------------------------------------------------------
// MLP layers 1+2 over row tiles of 16, layer 2 via v_wmma_f32_16x16x32_f16.
// Layer-1 weights live in per-lane registers laid out exactly as the WMMA
// A-fragment expects; W2 fragments (f16) are loop-invariant in VGPRs.
// C is preloaded with b2 broadcast so D = h1@W2 + b2 in one chained pair.
template <int X0MODE>
__global__ __launch_bounds__(256) void mlp_kernel(
    const float* __restrict__ x0arr, const float* __restrict__ x1arr,
    const float* __restrict__ W1, const float* __restrict__ b1,
    const float* __restrict__ W2, const float* __restrict__ b2,
    float* __restrict__ acc, int nTiles) {
  const int lane = threadIdx.x & 31;
  const int half = lane >> 4;      // K-half selector for A/B fragments
  const int ncol = lane & 15;      // N column / M row index
  const int waveId = blockIdx.x * (blockDim.x >> 5) + (threadIdx.x >> 5);
  const int nWaves = gridDim.x * (blockDim.x >> 5);

  // B fragments: W2 slice [K=32*kc .. +32) x [N=16*nt .. +16), ISA B layout:
  // VGPR j holds K = 2j+8h (j<4) or 16+2(j-4)+8h (j>=4), +{0,1}, N = lane&15.
  v16h bf[2][2];
#pragma unroll
  for (int nt = 0; nt < 2; ++nt)
#pragma unroll
    for (int kc = 0; kc < 2; ++kc)
#pragma unroll
      for (int j = 0; j < 8; ++j)
#pragma unroll
        for (int s = 0; s < 2; ++s) {
          int K = kc * 32 + ((j < 4) ? (2 * j + 8 * half) : (16 + 2 * (j - 4) + 8 * half)) + s;
          bf[nt][kc][j * 2 + s] = (_Float16)W2[K * 32 + nt * 16 + ncol];
        }

  // Per-lane layer-1 weight/bias slots matching the A-fragment K mapping:
  // VGPR j: K = 2j+8h (j<4) or 16+2(j-4)+8h (j>=4), two halves per VGPR.
  float wk1[32], wb[32], wk0[32];
#pragma unroll
  for (int kc = 0; kc < 2; ++kc)
#pragma unroll
    for (int j = 0; j < 8; ++j)
#pragma unroll
      for (int s = 0; s < 2; ++s) {
        int t = kc * 16 + j * 2 + s;
        int K = kc * 32 +
                ((j < 4) ? (2 * j + 8 * half) : (16 + 2 * (j - 4) + 8 * half)) + s;
        wk1[t] = W1[64 + K];
        wb[t] = b1[K];
        wk0[t] = X0MODE ? W1[K] : 0.0f;
      }

  const float c0i = b2[ncol], c1i = b2[16 + ncol];
  float sacc0[8], sacc1[8];
#pragma unroll
  for (int v = 0; v < 8; ++v) { sacc0[v] = 0.0f; sacc1[v] = 0.0f; }

  for (int tile = waveId; tile < nTiles; tile += nWaves) {
    size_t r = (size_t)tile * 16 + ncol;  // row handled by this lane (M=lane&15)
    float x1 = x1arr[r];
    float x0 = X0MODE ? x0arr[r] : 0.0f;

    // layer 1 + relu, packed straight into the A fragment as f16
    v16h alo, ahi;
#pragma unroll
    for (int kc = 0; kc < 2; ++kc)
#pragma unroll
      for (int j = 0; j < 8; ++j)
#pragma unroll
        for (int s = 0; s < 2; ++s) {
          int t = kc * 16 + j * 2 + s;
          float h = fmaf(x1, wk1[t], wb[t]);
          if (X0MODE) h = fmaf(x0, wk0[t], h);
          h = fmaxf(h, 0.0f);
          if (kc == 0) alo[j * 2 + s] = (_Float16)h;
          else         ahi[j * 2 + s] = (_Float16)h;
        }

    v8f d0, d1;
#pragma unroll
    for (int v = 0; v < 8; ++v) { d0[v] = c0i; d1[v] = c1i; }
    d0 = __builtin_amdgcn_wmma_f32_16x16x32_f16(false, alo, false, bf[0][0],
                                                (short)0, d0, false, false);
    d0 = __builtin_amdgcn_wmma_f32_16x16x32_f16(false, ahi, false, bf[0][1],
                                                (short)0, d0, false, false);
    d1 = __builtin_amdgcn_wmma_f32_16x16x32_f16(false, alo, false, bf[1][0],
                                                (short)0, d1, false, false);
    d1 = __builtin_amdgcn_wmma_f32_16x16x32_f16(false, ahi, false, bf[1][1],
                                                (short)0, d1, false, false);
#pragma unroll
    for (int v = 0; v < 8; ++v) {
      sacc0[v] += fmaxf(d0[v], 0.0f);  // relu(h2), summed over rows
      sacc1[v] += fmaxf(d1[v], 0.0f);
    }
  }

  // column sums: lane L and L+16 hold rows M=v and M=v+8 of column lane&15
  float p0s = 0.0f, p1s = 0.0f;
#pragma unroll
  for (int v = 0; v < 8; ++v) { p0s += sacc0[v]; p1s += sacc1[v]; }
  p0s += __shfl_xor(p0s, 16);
  p1s += __shfl_xor(p1s, 16);
  if (half == 0) {
    atomicAdd(&acc[ncol], p0s);
    atomicAdd(&acc[16 + ncol], p1s);
  }
}

// ---------------------------------------------------------------------------
// features = (acc0/count0 + acc1/count1) @ W3 + 2*b3
__global__ void feat_kernel(const float* __restrict__ ws,
                            const float* __restrict__ W3,
                            const float* __restrict__ b3,
                            float* __restrict__ out) {
  int f = blockIdx.x * blockDim.x + threadIdx.x;
  if (f >= FOUT) return;
  const float inv0 = 1.0f / (float)(B_BATCH * NM1);
  const float inv1 = 1.0f / (float)(B_BATCH * NF1);
  float s = 2.0f * b3[f];
#pragma unroll
  for (int k = 0; k < 32; ++k) {
    float mk = ws[OFF_ACC0 + k] * inv0 + ws[OFF_ACC1 + k] * inv1;
    s = fmaf(mk, W3[k * FOUT + f], s);
  }
  out[f] = s;
}

// ---------------------------------------------------------------------------
// entropy = -sum probs*log(probs+1e-10), separate normalization per diagram
__global__ __launch_bounds__(256) void entropy_kernel(float* __restrict__ ws) {
  __shared__ float red[8];
  int b = blockIdx.x * 256 + threadIdx.x;
  float S0 = fmaxf(ws[OFF_S + 0], 1e-30f);
  float S1 = fmaxf(ws[OFF_S + 1], 1e-30f);
  float e = 0.0f;
  float p = ws[OFF_P0 + b];
  if (p > 0.0f) { float t = p / S0; e -= t * logf(t + 1e-10f); }
  p = ws[OFF_Q1 + b];
  if (p > 0.0f) { float t = p / S1; e -= t * logf(t + 1e-10f); }
  p = ws[OFF_Q2 + b];
  if (p > 0.0f) { float t = p / S1; e -= t * logf(t + 1e-10f); }
  for (int off = 16; off; off >>= 1) e += __shfl_xor(e, off);
  if ((threadIdx.x & 31) == 0) red[threadIdx.x >> 5] = e;
  __syncthreads();
  if (threadIdx.x == 0) {
    float t = 0.0f;
    for (int i = 0; i < 8; ++i) t += red[i];
    atomicAdd(&ws[OFF_E], t);
  }
}

__global__ void write_entropy(const float* __restrict__ ws,
                              float* __restrict__ out) {
  out[0] = ws[OFF_E];
}

// ---------------------------------------------------------------------------
extern "C" void kernel_launch(void* const* d_in, const int* in_sizes, int n_in,
                              void* d_out, int out_size, void* d_ws,
                              size_t ws_size, hipStream_t stream) {
  (void)in_sizes; (void)n_in; (void)out_size; (void)ws_size;
  const float* pc = (const float*)d_in[0];
  const float* rb = (const float*)d_in[1];
  const float* rd = (const float*)d_in[2];
  const float* W1 = (const float*)d_in[3];
  const float* b1 = (const float*)d_in[4];
  const float* W2 = (const float*)d_in[5];
  const float* b2 = (const float*)d_in[6];
  const float* W3 = (const float*)d_in[7];
  const float* b3 = (const float*)d_in[8];
  float* out = (float*)d_out;   // [0,512)=features, [512,512+B)=euler, last=entropy
  float* ws = (float*)d_ws;

  hipLaunchKernelGGL(zero_kernel, dim3(1), dim3(128), 0, stream, ws);
  hipLaunchKernelGGL(geom_kernel, dim3(B_BATCH / 8), dim3(256), 0, stream,
                     pc, rb, rd, ws, out + FOUT);
  const int tiles0 = B_BATCH * NM1 / 16;  // 53248, exact
  const int tiles1 = B_BATCH * NF1 / 16;  // 12288, exact
  hipLaunchKernelGGL((mlp_kernel<0>), dim3(256), dim3(256), 0, stream,
                     (const float*)nullptr, ws + OFF_DEATHS0, W1, b1, W2, b2,
                     ws + OFF_ACC0, tiles0);
  hipLaunchKernelGGL((mlp_kernel<1>), dim3(64), dim3(256), 0, stream,
                     ws + OFF_B1, ws + OFF_D1, W1, b1, W2, b2,
                     ws + OFF_ACC1, tiles1);
  hipLaunchKernelGGL(feat_kernel, dim3(2), dim3(256), 0, stream, ws, W3, b3, out);
  hipLaunchKernelGGL(entropy_kernel, dim3(B_BATCH / 256), dim3(256), 0, stream, ws);
  hipLaunchKernelGGL(write_entropy, dim3(1), dim3(1), 0, stream, ws,
                     out + FOUT + B_BATCH);
}
